// TransformerDecoderLayer_32435593020030
// MI455X (gfx1250) — matrix operand order repaired and checked by reading, not compiled, not run on hardware
//
#include <hip/hip_runtime.h>
#include <hip/hip_bf16.h>

// ---------------------------------------------------------------------------
// Types for CDNA5 WMMA (gfx1250, wave32)
// ---------------------------------------------------------------------------
typedef __attribute__((ext_vector_type(16))) __bf16 v16bf;
typedef __attribute__((ext_vector_type(8)))  float  v8f;

union FragBF { v16bf v; unsigned u[8]; };

__device__ __forceinline__ unsigned short f32_to_bf16(float f) {
    unsigned u = __float_as_uint(f);
    unsigned r = u + 0x7FFFu + ((u >> 16) & 1u);   // round-nearest-even
    return (unsigned short)(r >> 16);
}

// A-fragment K offset for 16-bit A 16x32: VGPR j holds K pair 2j (+16*half).
__device__ __forceinline__ int a_koff(int j, int lh) {
    return 2 * j + 16 * lh;
}
// B-fragment K offset for 16-bit B 32x16: VGPR j holds K pair 2j (+16*half).
__device__ __forceinline__ int b_koff(int j, int lh) {
    return 2 * j + 16 * lh;
}

// CDNA5 async global->LDS copy (ASYNCcnt-tracked), 16B per lane.
#define ASYNC_CP_B128(ldsOff, gptr)                                           \
    asm volatile("global_load_async_to_lds_b128 %0, %1, off"                  \
                 :: "v"(ldsOff), "v"(gptr) : "memory")
#define WAIT_ASYNC0() asm volatile("s_wait_asynccnt 0x0" ::: "memory")

// ---------------------------------------------------------------------------
// f32 -> bf16 conversion (row-major copy, for activations)
// ---------------------------------------------------------------------------
__global__ void convert_bf16_kernel(const float* __restrict__ in,
                                    unsigned short* __restrict__ out, size_t n) {
    size_t i = (size_t)blockIdx.x * blockDim.x + threadIdx.x;
    if (i < n) out[i] = f32_to_bf16(in[i]);
}

// ---------------------------------------------------------------------------
// f32 [K][N] -> bf16 [N][K] tiled transpose (for weights; makes GEMM B-tiles
// K-contiguous so staging is pure B128 async copies).
// ---------------------------------------------------------------------------
__global__ __launch_bounds__(256)
void transpose_bf16_kernel(const float* __restrict__ in,
                           unsigned short* __restrict__ out, int K, int N) {
    __shared__ float tile[32][33];
    int kb = blockIdx.y * 32, nb = blockIdx.x * 32;
    int tx = threadIdx.x & 31, ty = threadIdx.x >> 5;   // 32 x 8
#pragma unroll
    for (int i = 0; i < 4; ++i)
        tile[ty + 8 * i][tx] = in[(size_t)(kb + ty + 8 * i) * N + nb + tx];
    __syncthreads();
#pragma unroll
    for (int i = 0; i < 4; ++i)
        out[(size_t)(nb + ty + 8 * i) * K + kb + tx] = f32_to_bf16(tile[tx][ty + 8 * i]);
}

// ---------------------------------------------------------------------------
// LayerNorm over D; writes optional f32 and optional bf16 outputs.
// ---------------------------------------------------------------------------
__global__ __launch_bounds__(256)
void layernorm_kernel(const float* __restrict__ x, const float* __restrict__ g,
                      const float* __restrict__ b, float* __restrict__ outF,
                      unsigned short* __restrict__ outB, int D) {
    __shared__ float red[256];
    int row = blockIdx.x, t = threadIdx.x;
    const float* xr = x + (size_t)row * D;
    float s = 0.f;
    for (int k = t; k < D; k += 256) s += xr[k];
    red[t] = s; __syncthreads();
    for (int off = 128; off; off >>= 1) { if (t < off) red[t] += red[t + off]; __syncthreads(); }
    float mean = red[0] / (float)D;
    __syncthreads();
    float v = 0.f;
    for (int k = t; k < D; k += 256) { float d = xr[k] - mean; v += d * d; }
    red[t] = v; __syncthreads();
    for (int off = 128; off; off >>= 1) { if (t < off) red[t] += red[t + off]; __syncthreads(); }
    float rstd = rsqrtf(red[0] / (float)D + 1e-6f);
    for (int k = t; k < D; k += 256) {
        float y = (xr[k] - mean) * rstd * g[k] + b[k];
        if (outF) outF[(size_t)row * D + k] = y;
        if (outB) outB[(size_t)row * D + k] = f32_to_bf16(y);
    }
}

// ---------------------------------------------------------------------------
// tau = sigmoid(LN(relu(x @ w + bias)))   w: [D,16]  -> tau [rows,16]
// ---------------------------------------------------------------------------
__global__ __launch_bounds__(256)
void tau_kernel(const float* __restrict__ x, const float* __restrict__ w,
                const float* __restrict__ bias, const float* __restrict__ g,
                const float* __restrict__ b2, float* __restrict__ tau, int D) {
    __shared__ float part[16][17];
    __shared__ float vals[16];
    int row = blockIdx.x, t = threadIdx.x;
    int j = t & 15, c = t >> 4;
    int chunk = D >> 4;
    const float* xr = x + (size_t)row * D;
    float s = 0.f;
    for (int k = c * chunk; k < (c + 1) * chunk; ++k) s += xr[k] * w[(size_t)k * 16 + j];
    part[c][j] = s;
    __syncthreads();
    if (t < 16) {
        float acc = 0.f;
        for (int cc = 0; cc < 16; ++cc) acc += part[cc][t];
        acc += bias[t];
        vals[t] = fmaxf(acc, 0.f);
    }
    __syncthreads();
    if (t < 16) {
        float m = 0.f;
        for (int i = 0; i < 16; ++i) m += vals[i];
        m *= (1.f / 16.f);
        float v = 0.f;
        for (int i = 0; i < 16; ++i) { float d = vals[i] - m; v += d * d; }
        v *= (1.f / 16.f);
        float y = (vals[t] - m) * rsqrtf(v + 1e-6f) * g[t] + b2[t];
        tau[(size_t)row * 16 + t] = 1.f / (1.f + __expf(-y));
    }
}

// ---------------------------------------------------------------------------
// Row softmax: out[row] = softmax(scores[row]) over Lc columns.
// ---------------------------------------------------------------------------
__global__ __launch_bounds__(256)
void softmax_kernel(const float* __restrict__ s, float* __restrict__ out, int Lc) {
    __shared__ float red[256];
    int row = blockIdx.x, t = threadIdx.x;
    const float* sr = s + (size_t)row * Lc;
    float m = -3.0e38f;
    for (int k = t; k < Lc; k += 256) m = fmaxf(m, sr[k]);
    red[t] = m; __syncthreads();
    for (int off = 128; off; off >>= 1) { if (t < off) red[t] = fmaxf(red[t], red[t + off]); __syncthreads(); }
    m = red[0]; __syncthreads();
    float sum = 0.f;
    for (int k = t; k < Lc; k += 256) sum += __expf(sr[k] - m);
    red[t] = sum; __syncthreads();
    for (int off = 128; off; off >>= 1) { if (t < off) red[t] += red[t + off]; __syncthreads(); }
    float inv = 1.f / red[0];
    for (int k = t; k < Lc; k += 256) out[(size_t)row * Lc + k] = __expf(sr[k] - m) * inv;
}

// ---------------------------------------------------------------------------
// bf16 GEMM:  out[M,N] = act(A[M,K] @ W[K,N] + bias (+ res)),  W given as
// WT[N][K] (pre-transposed).  Block 128x128, BLK_K=32, 8 waves (4x2), each
// wave owns 32x64 = 8 WMMAs per K-step.
// Double-buffered async global->LDS staging: copies for tile k+1 are issued
// before computing tile k, hiding copy latency behind the WMMA pipe.
// LDS row stride 40 halfwords: 80B (16B aligned for B128 async writes, and
// 20-bank stride => conflict-free fragment reads across the 16 lanes).
// ---------------------------------------------------------------------------
__global__ __launch_bounds__(256)
void gemm_bf16_kernel(const unsigned short* __restrict__ A,
                      const unsigned short* __restrict__ WT,
                      const float* __restrict__ bias,
                      const float* __restrict__ res,
                      float* __restrict__ outF,
                      unsigned short* __restrict__ outB,
                      int M, int N, int K, int relu) {
    __shared__ unsigned short sA[2][128][40];
    __shared__ unsigned short sB[2][128][40];
    int m0 = blockIdx.y * 128, n0 = blockIdx.x * 128;
    int t = threadIdx.x;
    int lane = t & 31, wid = t >> 5;
    int wm = wid >> 1, wn = wid & 1;        // 4 (M) x 2 (N) waves
    int lm = lane & 15, lh = lane >> 4;
    v8f acc[2][4] = {};

    // staging map: each thread copies 16 halfwords (two B128) per tile
    int srow = t >> 1, sseg = (t & 1) * 16;
    const unsigned short* aptr = A  + (size_t)(m0 + srow) * K + sseg;
    const unsigned short* bptr = WT + (size_t)(n0 + srow) * K + sseg;
    unsigned ldsA[2] = { (unsigned)(size_t)&sA[0][srow][sseg],
                         (unsigned)(size_t)&sA[1][srow][sseg] };
    unsigned ldsB[2] = { (unsigned)(size_t)&sB[0][srow][sseg],
                         (unsigned)(size_t)&sB[1][srow][sseg] };

    // prologue: stage tile 0 into buffer 0
    ASYNC_CP_B128(ldsA[0],      aptr);
    ASYNC_CP_B128(ldsA[0] + 16, aptr + 8);
    ASYNC_CP_B128(ldsB[0],      bptr);
    ASYNC_CP_B128(ldsB[0] + 16, bptr + 8);
    aptr += 32; bptr += 32;
    WAIT_ASYNC0();
    __syncthreads();

    int nsteps = K >> 5;
    for (int s = 0; s < nsteps; ++s) {
        int cur = s & 1, nxt = cur ^ 1;
        if (s + 1 < nsteps) {
            // issue copies for the next tile; the "nxt" buffer was fully
            // consumed by all waves before the barrier that ended step s-1.
            ASYNC_CP_B128(ldsA[nxt],      aptr);
            ASYNC_CP_B128(ldsA[nxt] + 16, aptr + 8);
            ASYNC_CP_B128(ldsB[nxt],      bptr);
            ASYNC_CP_B128(ldsB[nxt] + 16, bptr + 8);
            __builtin_prefetch(aptr + 32, 0, 1);   // L2 prefetch one tile ahead
            __builtin_prefetch(bptr + 32, 0, 1);
            aptr += 32; bptr += 32;
        }

        FragBF a0, a1, b[4];
#pragma unroll
        for (int j = 0; j < 8; ++j) {
            int ka = a_koff(j, lh);
            a0.u[j] = *(const unsigned*)&sA[cur][wm * 32 + lm][ka];
            a1.u[j] = *(const unsigned*)&sA[cur][wm * 32 + 16 + lm][ka];
            int kb = b_koff(j, lh);
#pragma unroll
            for (int bn = 0; bn < 4; ++bn)
                b[bn].u[j] = *(const unsigned*)&sB[cur][wn * 64 + bn * 16 + lm][kb];
        }
#pragma unroll
        for (int bn = 0; bn < 4; ++bn) {
            acc[0][bn] = __builtin_amdgcn_wmma_f32_16x16x32_bf16(false, a0.v, false, b[bn].v,
                                                                 (short)0, acc[0][bn], false, false);
            acc[1][bn] = __builtin_amdgcn_wmma_f32_16x16x32_bf16(false, a1.v, false, b[bn].v,
                                                                 (short)0, acc[1][bn], false, false);
        }
        WAIT_ASYNC0();       // next tile landed (own wave's copies)
        __syncthreads();     // all waves done reading `cur` + copies visible
    }

    // epilogue: C/D layout -> row = r + 8*lh within 16-tile, col = lane%16
    float bv[4];
#pragma unroll
    for (int bn = 0; bn < 4; ++bn)
        bv[bn] = bias ? bias[n0 + wn * 64 + bn * 16 + lm] : 0.f;
#pragma unroll
    for (int am = 0; am < 2; ++am) {
#pragma unroll
        for (int bn = 0; bn < 4; ++bn) {
            int col = n0 + wn * 64 + bn * 16 + lm;
#pragma unroll
            for (int r = 0; r < 8; ++r) {
                int row = m0 + wm * 32 + am * 16 + 8 * lh + r;
                float v = acc[am][bn][r] + bv[bn];
                if (res)  v += res[(size_t)row * N + col];
                if (relu) v = fmaxf(v, 0.f);
                if (outF) outF[(size_t)row * N + col] = v;
                if (outB) outB[(size_t)row * N + col] = f32_to_bf16(v);
            }
        }
    }
}

// ---------------------------------------------------------------------------
// Flash attention, one wave per (b, h, 16-query tile), dh = 64.
// Q/K/V bf16 laid out [B*L, D] with head offset h*64.
// tau: [B*Lq, H].  mask: [B, Lk] bool (pad).  scoreO (head 0 only, cross):
// raw masked/scaled scores [B, Lq, Lk] feeding the returned attn output.
// ---------------------------------------------------------------------------
__global__ __launch_bounds__(32)
void flash_attn_kernel(const unsigned short* __restrict__ Qb,
                       const unsigned short* __restrict__ Kb,
                       const unsigned short* __restrict__ Vb,
                       const float* __restrict__ tau,
                       const unsigned char* __restrict__ mask,
                       unsigned short* __restrict__ ctxO,
                       float* __restrict__ scoreO,
                       int Lq, int Lk, int Dm, int causal) {
    const int DH = 64;
    __shared__ float stau[16];
    __shared__ unsigned short sP[16][34];
    int qt = blockIdx.x, h = blockIdx.y, b = blockIdx.z;
    int Hn = gridDim.y;
    int q0 = qt * 16;
    int lane = threadIdx.x, lm = lane & 15, lh = lane >> 4;

    size_t qbase  = (size_t)(b * Lq + q0) * Dm + h * DH;
    size_t kvbase = (size_t)b * Lk * Dm + h * DH;

    if (lane < 16) stau[lane] = tau[(size_t)(b * Lq + q0 + lane) * Hn + h];
    float tsc[8];
#pragma unroll
    for (int r = 0; r < 8; ++r) tsc[r] = stau[8 * lh + r] * 0.125f;  // tau / sqrt(64)

    FragBF qf0, qf1;
#pragma unroll
    for (int j = 0; j < 8; ++j) {
        int ka = a_koff(j, lh);
        qf0.u[j] = *(const unsigned*)(Qb + qbase + (size_t)lm * Dm + ka);
        qf1.u[j] = *(const unsigned*)(Qb + qbase + (size_t)lm * Dm + 32 + ka);
    }

    float mrow[8], lrow[8];
    v8f C0 = {}, C1 = {}, C2 = {}, C3 = {};
#pragma unroll
    for (int r = 0; r < 8; ++r) { mrow[r] = -3.0e38f; lrow[r] = 0.f; }

    int ktEnd = causal ? (q0 + 16) : Lk;   // no fully-masked key tile under causal
    for (int kt = 0; kt < ktEnd; kt += 32) {
        FragBF kf; v8f S0 = {}, S1 = {};
        const unsigned short* k0p = Kb + kvbase + (size_t)(kt + lm) * Dm;
        const unsigned short* k1p = Kb + kvbase + (size_t)(kt + 16 + lm) * Dm;
#pragma unroll
        for (int j = 0; j < 8; ++j) kf.u[j] = *(const unsigned*)(k0p + b_koff(j, lh));
        S0 = __builtin_amdgcn_wmma_f32_16x16x32_bf16(false, qf0.v, false, kf.v, (short)0, S0, false, false);
#pragma unroll
        for (int j = 0; j < 8; ++j) kf.u[j] = *(const unsigned*)(k0p + 32 + b_koff(j, lh));
        S0 = __builtin_amdgcn_wmma_f32_16x16x32_bf16(false, qf1.v, false, kf.v, (short)0, S0, false, false);
#pragma unroll
        for (int j = 0; j < 8; ++j) kf.u[j] = *(const unsigned*)(k1p + b_koff(j, lh));
        S1 = __builtin_amdgcn_wmma_f32_16x16x32_bf16(false, qf0.v, false, kf.v, (short)0, S1, false, false);
#pragma unroll
        for (int j = 0; j < 8; ++j) kf.u[j] = *(const unsigned*)(k1p + 32 + b_koff(j, lh));
        S1 = __builtin_amdgcn_wmma_f32_16x16x32_bf16(false, qf1.v, false, kf.v, (short)0, S1, false, false);

        int key0 = kt + lm, key1 = kt + 16 + lm;
        bool pm0 = mask && mask[(size_t)b * Lk + key0];
        bool pm1 = mask && mask[(size_t)b * Lk + key1];

#pragma unroll
        for (int r = 0; r < 8; ++r) {
            int qrow = q0 + 8 * lh + r;
            float s0 = S0[r] * tsc[r];
            float s1 = S1[r] * tsc[r];
            if (pm0 || (causal && key0 > qrow)) s0 = -1.0e30f;
            if (pm1 || (causal && key1 > qrow)) s1 = -1.0e30f;
            if (scoreO && h == 0) {
                scoreO[(size_t)(b * Lq + qrow) * Lk + key0] = s0;
                scoreO[(size_t)(b * Lq + qrow) * Lk + key1] = s1;
            }
            float mx = fmaxf(s0, s1);   // row lives within a 16-lane half
            mx = fmaxf(mx, __shfl_xor(mx, 1)); mx = fmaxf(mx, __shfl_xor(mx, 2));
            mx = fmaxf(mx, __shfl_xor(mx, 4)); mx = fmaxf(mx, __shfl_xor(mx, 8));
            float mnew = fmaxf(mrow[r], mx);
            float corr = __expf(mrow[r] - mnew);
            float p0 = __expf(s0 - mnew), p1 = __expf(s1 - mnew);
            float rs = p0 + p1;
            rs += __shfl_xor(rs, 1); rs += __shfl_xor(rs, 2);
            rs += __shfl_xor(rs, 4); rs += __shfl_xor(rs, 8);
            lrow[r] = lrow[r] * corr + rs;
            mrow[r] = mnew;
            C0[r] *= corr; C1[r] *= corr; C2[r] *= corr; C3[r] *= corr;
            sP[8 * lh + r][lm] = f32_to_bf16(p0);
            sP[8 * lh + r][16 + lm] = f32_to_bf16(p1);
        }

        FragBF pf;   // re-layout P (C layout -> A fragment) via LDS
#pragma unroll
        for (int j = 0; j < 8; ++j) pf.u[j] = *(const unsigned*)&sP[lm][a_koff(j, lh)];

#pragma unroll
        for (int tdh = 0; tdh < 4; ++tdh) {   // ctx += P(16x32) @ V(32x64)
            FragBF vf;
#pragma unroll
            for (int j = 0; j < 8; ++j) {
                int keyl = kt + 16 * lh + 2 * j;
                const unsigned short* vp = Vb + kvbase + (size_t)keyl * Dm + tdh * 16 + lm;
                unsigned lo = *vp, hi = *(vp + Dm);
                vf.u[j] = lo | (hi << 16);
            }
            v8f* C = (tdh == 0) ? &C0 : (tdh == 1) ? &C1 : (tdh == 2) ? &C2 : &C3;
            *C = __builtin_amdgcn_wmma_f32_16x16x32_bf16(false, pf.v, false, vf.v, (short)0, *C, false, false);
        }
    }

#pragma unroll
    for (int r = 0; r < 8; ++r) {
        float inv = lrow[r] > 0.f ? 1.f / lrow[r] : 0.f;
        size_t obase = (size_t)(b * Lq + q0 + 8 * lh + r) * Dm + h * DH;
        ctxO[obase + 0 + lm]  = f32_to_bf16(C0[r] * inv);
        ctxO[obase + 16 + lm] = f32_to_bf16(C1[r] * inv);
        ctxO[obase + 32 + lm] = f32_to_bf16(C2[r] * inv);
        ctxO[obase + 48 + lm] = f32_to_bf16(C3[r] * inv);
    }
}

// ---------------------------------------------------------------------------
// Host orchestration
// ---------------------------------------------------------------------------
extern "C" void kernel_launch(void* const* d_in, const int* in_sizes, int n_in,
                              void* d_out, int out_size, void* d_ws, size_t ws_size,
                              hipStream_t stream) {
    const int Bb = 4, LT = 1024, LS = 1024, D = 1024, H = 16, DFF = 4096;
    const int M = Bb * LT;

    const float* inputs  = (const float*)d_in[0];
    const float* memory  = (const float*)d_in[1];
    const unsigned char* srcMask = (const unsigned char*)d_in[2];
    const unsigned char* tgtMask = (const unsigned char*)d_in[3];
    const float* ln1g = (const float*)d_in[4];  const float* ln1b = (const float*)d_in[5];
    const float* ln2g = (const float*)d_in[6];  const float* ln2b = (const float*)d_in[7];
    const float* sa_wq = (const float*)d_in[8];  const float* sa_bq = (const float*)d_in[9];
    const float* sa_wk = (const float*)d_in[10]; const float* sa_bk = (const float*)d_in[11];
    const float* sa_wv = (const float*)d_in[12]; const float* sa_bv = (const float*)d_in[13];
    const float* sa_wo = (const float*)d_in[14]; const float* sa_bo = (const float*)d_in[15];
    const float* ca_wq = (const float*)d_in[16]; const float* ca_bq = (const float*)d_in[17];
    const float* ca_wk = (const float*)d_in[18]; const float* ca_bk = (const float*)d_in[19];
    const float* ca_wv = (const float*)d_in[20]; const float* ca_bv = (const float*)d_in[21];
    const float* ca_wo = (const float*)d_in[22]; const float* ca_bo = (const float*)d_in[23];
    const float* slw = (const float*)d_in[24]; const float* slb = (const float*)d_in[25];
    const float* slg = (const float*)d_in[26]; const float* slb2 = (const float*)d_in[27];
    const float* clw = (const float*)d_in[28]; const float* clb = (const float*)d_in[29];
    const float* clg = (const float*)d_in[30]; const float* clb2 = (const float*)d_in[31];
    const float* fng = (const float*)d_in[32]; const float* fnb = (const float*)d_in[33];
    const float* w1 = (const float*)d_in[34]; const float* b1 = (const float*)d_in[35];
    const float* w2 = (const float*)d_in[36]; const float* b2 = (const float*)d_in[37];

    float* outO     = (float*)d_out;                       // [B,LT,D]
    float* outAttn  = outO + (size_t)M * D;                // [B,LT,LS]
    float* outINorm = outAttn + (size_t)Bb * LT * LS;      // [B,LT,D]

    char* wsb = (char*)d_ws;
    size_t off = 0;
    auto alloc = [&](size_t bytes) -> void* {
        void* p = wsb + off;
        off = (off + bytes + 255) & ~(size_t)255;
        return p;
    };
    unsigned short* wsaq = (unsigned short*)alloc((size_t)D * D * 2);   // all weights
    unsigned short* wsak = (unsigned short*)alloc((size_t)D * D * 2);   // stored [N][K]
    unsigned short* wsav = (unsigned short*)alloc((size_t)D * D * 2);
    unsigned short* wsao = (unsigned short*)alloc((size_t)D * D * 2);
    unsigned short* wcaq = (unsigned short*)alloc((size_t)D * D * 2);
    unsigned short* wcak = (unsigned short*)alloc((size_t)D * D * 2);
    unsigned short* wcav = (unsigned short*)alloc((size_t)D * D * 2);
    unsigned short* wcao = (unsigned short*)alloc((size_t)D * D * 2);
    unsigned short* w1b  = (unsigned short*)alloc((size_t)D * DFF * 2); // [DFF][D]
    unsigned short* w2b  = (unsigned short*)alloc((size_t)DFF * D * 2); // [D][DFF]
    unsigned short* inormB = (unsigned short*)alloc((size_t)M * D * 2); // reused as xlnB
    unsigned short* qB   = (unsigned short*)alloc((size_t)M * D * 2);   // self then cross
    unsigned short* kB   = (unsigned short*)alloc((size_t)M * D * 2);
    unsigned short* vB   = (unsigned short*)alloc((size_t)M * D * 2);
    unsigned short* ctxB = (unsigned short*)alloc((size_t)M * D * 2);   // self then cross
    float* queryF = (float*)alloc((size_t)M * D * 4);
    float* qnF    = (float*)alloc((size_t)M * D * 4);
    unsigned short* qnB = (unsigned short*)alloc((size_t)M * D * 2);
    float* tauS   = (float*)alloc((size_t)M * H * 4);
    float* tauC   = (float*)alloc((size_t)M * H * 4);
    unsigned short* memB = (unsigned short*)alloc((size_t)Bb * LS * D * 2);
    float* scoreF = (float*)alloc((size_t)Bb * LT * LS * 4);
    float* xF     = (float*)alloc((size_t)M * D * 4);
    unsigned short* interB = (unsigned short*)alloc((size_t)M * DFF * 2);

    auto conv = [&](const float* src, unsigned short* dst, size_t n) {
        convert_bf16_kernel<<<(unsigned)((n + 255) / 256), 256, 0, stream>>>(src, dst, n);
    };
    auto tconv = [&](const float* src, unsigned short* dst, int Kk, int Nn) {
        dim3 g(Nn / 32, Kk / 32);
        transpose_bf16_kernel<<<g, 256, 0, stream>>>(src, dst, Kk, Nn);
    };
    auto gemm = [&](const unsigned short* A, const unsigned short* Wm, const float* bias,
                    const float* res, float* oF, unsigned short* oB, int Mm, int Nn, int Kk, int relu) {
        dim3 g(Nn / 128, Mm / 128);
        gemm_bf16_kernel<<<g, 256, 0, stream>>>(A, Wm, bias, res, oF, oB, Mm, Nn, Kk, relu);
    };

    // ---- weight transpose-conversions ([K][N] f32 -> [N][K] bf16) ----
    tconv(sa_wq, wsaq, D, D); tconv(sa_wk, wsak, D, D);
    tconv(sa_wv, wsav, D, D); tconv(sa_wo, wsao, D, D);
    tconv(ca_wq, wcaq, D, D); tconv(ca_wk, wcak, D, D);
    tconv(ca_wv, wcav, D, D); tconv(ca_wo, wcao, D, D);
    tconv(w1, w1b, D, DFF);   tconv(w2, w2b, DFF, D);
    conv(memory, memB, (size_t)Bb * LS * D);

    // ---- input_norm = LN(inputs); also third output ----
    layernorm_kernel<<<M, 256, 0, stream>>>(inputs, ln1g, ln1b, outINorm, inormB, D);

    // ---- tau_self ----
    tau_kernel<<<M, 256, 0, stream>>>(outINorm, slw, slb, slg, slb2, tauS, D);

    // ---- self-attention QKV projections ----
    gemm(inormB, wsaq, sa_bq, nullptr, nullptr, qB, M, D, D, 0);
    gemm(inormB, wsak, sa_bk, nullptr, nullptr, kB, M, D, D, 0);
    gemm(inormB, wsav, sa_bv, nullptr, nullptr, vB, M, D, D, 0);

    // ---- self flash attention (causal + tgt pad mask) ----
    {
        dim3 g(LT / 16, H, Bb);
        flash_attn_kernel<<<g, 32, 0, stream>>>(qB, kB, vB, tauS, tgtMask, ctxB, nullptr,
                                                LT, LT, D, 1);
    }

    // ---- self out-proj + residual -> query ----
    gemm(ctxB, wsao, sa_bo, inputs, queryF, nullptr, M, D, D, 0);

    // ---- query_norm = LN(query) ----
    layernorm_kernel<<<M, 256, 0, stream>>>(queryF, ln2g, ln2b, qnF, qnB, D);

    // ---- tau_ctxt ----
    tau_kernel<<<M, 256, 0, stream>>>(qnF, clw, clb, clg, clb2, tauC, D);

    // ---- cross-attention QKV ----
    gemm(qnB,  wcaq, ca_bq, nullptr, nullptr, qB, M, D, D, 0);
    gemm(memB, wcak, ca_bk, nullptr, nullptr, kB, M, D, D, 0);
    gemm(memB, wcav, ca_bv, nullptr, nullptr, vB, M, D, D, 0);

    // ---- cross flash attention; head-0 scores kept for attn output ----
    {
        dim3 g(LT / 16, H, Bb);
        flash_attn_kernel<<<g, 32, 0, stream>>>(qB, kB, vB, tauC, srcMask, ctxB, scoreF,
                                                LT, LS, D, 0);
    }

    // ---- attn output = softmax over stored head-0 scores ----
    softmax_kernel<<<Bb * LT, 256, 0, stream>>>(scoreF, outAttn, LS);

    // ---- cross out-proj + residual -> x ----
    gemm(ctxB, wcao, ca_bo, queryF, xF, nullptr, M, D, D, 0);

    // ---- FFN: LN -> relu-gemm -> gemm + residual ----
    layernorm_kernel<<<M, 256, 0, stream>>>(xF, fng, fnb, nullptr, inormB, D);
    gemm(inormB, w1b, b1, nullptr, nullptr, interB, M, DFF, D, 1);
    gemm(interB, w2b, b2, xF, outO, nullptr, M, D, DFF, 0);
}
